// Transformer_30777735644023
// MI455X (gfx1250) — compile-verified, matrix operand order checked by reading, not run on hardware
//
#include <hip/hip_runtime.h>
#include <hip/hip_bf16.h>
#include <math.h>

// ---------------- problem constants (from reference) ----------------
#define B_    4
#define M_    2048
#define C_    1024
#define H_    8
#define DQK_  32
#define DV_   128
#define NQ_   256            // C/4
#define MT_   (B_*M_)        // 8192 total rows
#define LOG2E 1.44269504088896340736f
#define SCALE_QK 0.17677669529663687f   // 1/sqrt(32)
#define LEAKY 0.2f
#define KSTEP 64             // keys per attention inner iteration

typedef __attribute__((ext_vector_type(16))) __bf16 v16bf;
typedef __attribute__((ext_vector_type(8)))  __bf16 v8bf;
typedef __attribute__((ext_vector_type(8)))  float  v8f;

// Build a 16x32 (A) or 32x16 (B) bf16 WMMA fragment from a row pointer.
// Per ISA layout: lane l, element e -> k = (e&7) + 8*(2*(e>>3) + (l>>4)).
// So each lane needs two contiguous 8-half chunks at k0 = 8*hl and 16+8*hl.
__device__ __forceinline__ v16bf frag_ab(const __bf16* rowptr, int hl) {
  v8bf a = *(const v8bf*)(rowptr + 8*hl);
  v8bf b = *(const v8bf*)(rowptr + 16 + 8*hl);
  v16bf r;
#pragma unroll
  for (int i = 0; i < 8; ++i) { r[i] = a[i]; r[8+i] = b[i]; }
  return r;
}

// ---------------- prep kernels ----------------
__global__ void k_cvt_bf16(const float* __restrict__ in, __bf16* __restrict__ out, int n) {
  int i = blockIdx.x * blockDim.x + threadIdx.x;
  if (i < n) out[i] = (__bf16)in[i];
}

// W[K x N] fp32 row-major -> WT[N x K] bf16 (so B-fragments load contiguously)
__global__ void k_transpose_bf16(const float* __restrict__ W, __bf16* __restrict__ WT,
                                 int K, int N) {
  int i = blockIdx.x * blockDim.x + threadIdx.x;
  if (i < K * N) {
    int k = i / N, n = i % N;
    WT[(size_t)n * K + k] = (__bf16)W[i];
  }
}

// ---------------- projection GEMM: Out = A[M x K] @ WT[N x K]^T + bias ----------------
// One wave computes a 16x64 strip: 1 A-fragment reused by 4 B-fragments per K-step.
// vmode=0: Out[m][n] row-major (Q/K projections).
// vmode=1: V projection, stored pre-transposed per head: Out[((b*H+h)*DV+dv)*M + m2]
//          so the attention kernel can load V B-fragments straight from global.
__global__ __launch_bounds__(128) void k_gemm_bias(
    const __bf16* __restrict__ A, const __bf16* __restrict__ WT,
    const float* __restrict__ bias, __bf16* __restrict__ Out,
    int Mrows, int N, int K, int vmode)
{
  const int lane = threadIdx.x & 31;
  const int wave = threadIdx.x >> 5;
  const int hl   = lane >> 4;
  const int lo   = lane & 15;
  const int rowTiles = Mrows >> 4;
  const int t     = blockIdx.x * 4 + wave;
  const int strip = t / rowTiles;
  const int rtile = t % rowTiles;
  const int row0  = rtile * 16;
  const int col0  = strip * 64;

  v8f acc[4] = {};
  const __bf16* aRow = A  + (size_t)(row0 + lo) * K;
  const __bf16* bRow = WT + (size_t)(col0 + lo) * K;

  for (int k0 = 0; k0 < K; k0 += 32) {
    v16bf af = frag_ab(aRow + k0, hl);
#pragma unroll
    for (int j = 0; j < 4; ++j) {
      v16bf wf = frag_ab(bRow + (size_t)(16*j) * K + k0, hl);
      acc[j] = __builtin_amdgcn_wmma_f32_16x16x32_bf16(
          false, af, false, wf, (short)0, acc[j], false, false);
    }
  }
#pragma unroll
  for (int j = 0; j < 4; ++j) {
    int n = col0 + 16*j + lo;
    float bb = bias[n];
#pragma unroll
    for (int r = 0; r < 8; ++r) {
      int m = row0 + r + 8*hl;
      __bf16 val = (__bf16)(acc[j][r] + bb);
      if (vmode) {
        int bb_i = m >> 11;            // m / M_
        int m2   = m & (M_ - 1);
        int hh   = n >> 7;             // n / DV_
        int dv   = n & (DV_ - 1);
        Out[(((size_t)bb_i * H_ + hh) * DV_ + dv) * M_ + m2] = val;
      } else {
        Out[(size_t)m * N + n] = val;
      }
    }
  }
}

// ---------------- flash attention + residual ----------------
// grid: (M/64, B*H); block: 128 threads = 4 waves, each wave owns 16 queries.
// No inter-wave communication: V is pre-transposed in global, P goes through a
// small per-wave LDS tile only (C-layout -> A-layout conversion).
__global__ __launch_bounds__(128) void k_attn(
    const __bf16* __restrict__ Qb, const __bf16* __restrict__ Kb,
    const __bf16* __restrict__ Vt, const float* __restrict__ x,
    const float* __restrict__ gamma_p, float* __restrict__ out)
{
  __shared__ __attribute__((aligned(16))) __bf16 ldsP[4][16 * 80]; // per-wave P: [row][key], stride 80

  const int tid  = threadIdx.x;
  const int lane = tid & 31;
  const int wave = tid >> 5;
  const int hl   = lane >> 4;
  const int lo   = lane & 15;

  const int b  = blockIdx.y / H_;
  const int h  = blockIdx.y % H_;
  const int q0 = blockIdx.x * 64 + wave * 16;
  const size_t bM = (size_t)b * M_;

  // Q A-fragment: 16 queries x DQK(=32) -- exactly one WMMA depth, loaded once.
  const __bf16* qptr = Qb + (bM + q0 + lo) * NQ_ + h * DQK_;
  v16bf qf = frag_ab(qptr, hl);

  const __bf16* vhead = Vt + ((size_t)(b * H_ + h) * DV_) * M_;  // [dv][key] for this head
  const __bf16* khead = Kb + bM * NQ_ + h * DQK_;                 // row stride NQ_

  v8f O[8] = {};
  float mrow[8], lrow[8];
#pragma unroll
  for (int r = 0; r < 8; ++r) { mrow[r] = -INFINITY; lrow[r] = 0.f; }
  const float gamma = gamma_p[0];

  __bf16* pw = &ldsP[wave][0];

  for (int kt = 0; kt < M_; kt += KSTEP) {
    // prefetch next tile's K rows and V rows into cache (global_prefetch_b8)
    if (kt + KSTEP < M_) {
      __builtin_prefetch(khead + (size_t)(kt + KSTEP + lane) * NQ_, 0, 1);
#pragma unroll
      for (int c = 0; c < 4; ++c)
        __builtin_prefetch(vhead + (size_t)(lane + 32 * c) * M_ + kt + KSTEP, 0, 1);
    }

    // S = Q * K^T for four 16-key subtiles (K B-fragments straight from global)
    v8f sc[4];
#pragma unroll
    for (int st = 0; st < 4; ++st) {
      v16bf kf = frag_ab(khead + (size_t)(kt + 16 * st + lo) * NQ_, hl);
      v8f z = {};
      sc[st] = __builtin_amdgcn_wmma_f32_16x16x32_bf16(
          false, qf, false, kf, (short)0, z, false, false);
    }

    // online softmax, row-wise (row r lives in VGPR r; N striped over 16-lane halves)
#pragma unroll
    for (int r = 0; r < 8; ++r) {
      float a[4];
#pragma unroll
      for (int st = 0; st < 4; ++st) {
        float v = sc[st][r] * SCALE_QK;
        a[st] = v >= 0.f ? v : LEAKY * v;
      }
      float rmax = fmaxf(fmaxf(a[0], a[1]), fmaxf(a[2], a[3]));
#pragma unroll
      for (int sh = 1; sh < 16; sh <<= 1) rmax = fmaxf(rmax, __shfl_xor(rmax, sh, 32));
      float mnew = fmaxf(mrow[r], rmax);
      float fac  = exp2f((mrow[r] - mnew) * LOG2E);
      float p[4], ps = 0.f;
#pragma unroll
      for (int st = 0; st < 4; ++st) { p[st] = exp2f((a[st] - mnew) * LOG2E); ps += p[st]; }
#pragma unroll
      for (int sh = 1; sh < 16; sh <<= 1) ps += __shfl_xor(ps, sh, 32);
      lrow[r] = lrow[r] * fac + ps;
      mrow[r] = mnew;
#pragma unroll
      for (int j = 0; j < 8; ++j) O[j][r] *= fac;
      int rr = r + 8 * hl;
#pragma unroll
      for (int st = 0; st < 4; ++st)
        pw[rr * 80 + 16 * st + lo] = (__bf16)p[st];
    }
    asm volatile("" ::: "memory");  // keep P stores before P-fragment reads (same-wave LDS is in-order)

    // O += P(16x64) * V(64x128): two 32-deep halves, V B-frags direct from global
#pragma unroll
    for (int ks = 0; ks < 2; ++ks) {
      v16bf pf = frag_ab(pw + lo * 80 + 32 * ks, hl);
#pragma unroll
      for (int j = 0; j < 8; ++j) {
        v16bf vf = frag_ab(vhead + (size_t)(16 * j + lo) * M_ + kt + 32 * ks, hl);
        O[j] = __builtin_amdgcn_wmma_f32_16x16x32_bf16(
            false, pf, false, vf, (short)0, O[j], false, false);
      }
    }
  }

  // epilogue: O/l * gamma + x  -> fp32 out
#pragma unroll
  for (int r = 0; r < 8; ++r) {
    int m = q0 + r + 8 * hl;
    float invl = 1.0f / lrow[r];
    size_t base = (bM + m) * (size_t)C_ + h * DV_;
#pragma unroll
    for (int j = 0; j < 8; ++j) {
      int col = 16 * j + lo;
      out[base + col] = gamma * O[j][r] * invl + x[base + col];
    }
  }
}

// ---------------- launcher ----------------
extern "C" void kernel_launch(void* const* d_in, const int* in_sizes, int n_in,
                              void* d_out, int out_size, void* d_ws, size_t ws_size,
                              hipStream_t stream) {
  (void)in_sizes; (void)n_in; (void)out_size; (void)ws_size;
  const float* x     = (const float*)d_in[0];
  const float* y     = (const float*)d_in[1];
  const float* Wq    = (const float*)d_in[2];
  const float* bq    = (const float*)d_in[3];
  const float* Wk    = (const float*)d_in[4];
  const float* bk    = (const float*)d_in[5];
  const float* Wv    = (const float*)d_in[6];
  const float* bv    = (const float*)d_in[7];
  const float* gamma = (const float*)d_in[8];
  float* out = (float*)d_out;

  // workspace carve-up (~62 MB total)
  char* ws = (char*)d_ws;
  __bf16* Xbf = (__bf16*)ws; ws += (size_t)MT_ * C_  * 2;
  __bf16* Ybf = (__bf16*)ws; ws += (size_t)MT_ * C_  * 2;
  __bf16* WqT = (__bf16*)ws; ws += (size_t)NQ_ * C_  * 2;
  __bf16* WkT = (__bf16*)ws; ws += (size_t)NQ_ * C_  * 2;
  __bf16* WvT = (__bf16*)ws; ws += (size_t)C_  * C_  * 2;
  __bf16* Qb  = (__bf16*)ws; ws += (size_t)MT_ * NQ_ * 2;
  __bf16* Kb  = (__bf16*)ws; ws += (size_t)MT_ * NQ_ * 2;
  __bf16* Vt  = (__bf16*)ws; ws += (size_t)MT_ * C_  * 2;   // per-head transposed V

  const int n = MT_ * C_;
  k_cvt_bf16<<<(n + 255) / 256, 256, 0, stream>>>(x, Xbf, n);
  k_cvt_bf16<<<(n + 255) / 256, 256, 0, stream>>>(y, Ybf, n);
  k_transpose_bf16<<<(C_ * NQ_ + 255) / 256, 256, 0, stream>>>(Wq, WqT, C_, NQ_);
  k_transpose_bf16<<<(C_ * NQ_ + 255) / 256, 256, 0, stream>>>(Wk, WkT, C_, NQ_);
  k_transpose_bf16<<<(C_ * C_  + 255) / 256, 256, 0, stream>>>(Wv, WvT, C_, C_);

  {
    int tiles = (MT_ / 16) * (NQ_ / 64);   // 2048 wave-tiles
    k_gemm_bias<<<tiles / 4, 128, 0, stream>>>(Xbf, WqT, bq, Qb, MT_, NQ_, C_, 0);
    k_gemm_bias<<<tiles / 4, 128, 0, stream>>>(Ybf, WkT, bk, Kb, MT_, NQ_, C_, 0);
  }
  {
    int tiles = (MT_ / 16) * (C_ / 64);    // 8192 wave-tiles
    k_gemm_bias<<<tiles / 4, 128, 0, stream>>>(Ybf, WvT, bv, Vt, MT_, C_, C_, 1);
  }

  dim3 g(M_ / 64, B_ * H_);
  k_attn<<<g, 128, 0, stream>>>(Qb, Kb, Vt, x, gamma, out);
}
